// TemporalGCN_22608707846704
// MI455X (gfx1250) — matrix operand order repaired and checked by reading, not compiled, not run on hardware
//
#include <hip/hip_runtime.h>
#include <hip/hip_bf16.h>

typedef __attribute__((ext_vector_type(16))) __bf16 v16bf;
typedef __attribute__((ext_vector_type(8)))  __bf16 v8bf;
typedef __attribute__((ext_vector_type(8)))  float  v8f;

namespace cfg {
constexpr int B    = 128;
constexpr int CIN  = 9;
constexpr int T    = 2048;
constexpr int C1   = 16;
constexpr int T1   = 1024;   // after pool1
constexpr int C2   = 32;     // FEAT
constexpr int NEWT = 512;    // after pool2
constexpr int N    = B * NEWT;   // 65536 nodes
constexpr int E    = 524288;
constexpr int HID  = 128;
constexpr int OUT  = 6;
}

// ---------------------------------------------------------------------------
// Stage 1: conv1d(9->16,k5,same) + relu + maxpool2.  h1 layout [B][C1][T1].
// idx = ((b*C1)+co)*T1 + tp  -> per-wave-uniform weight indices, coalesced x.
// ---------------------------------------------------------------------------
__global__ void conv1_relu_pool(const float* __restrict__ x,
                                const float* __restrict__ w,
                                const float* __restrict__ bias,
                                float* __restrict__ h1) {
  int idx = blockIdx.x * blockDim.x + threadIdx.x;     // B*C1*T1 = 2,097,152
  int tp = idx & (cfg::T1 - 1);
  int co = (idx >> 10) & (cfg::C1 - 1);
  int bb = idx >> 14;
  const float* xb = x + (size_t)bb * cfg::CIN * cfg::T;
  const float* wc = w + (size_t)co * cfg::CIN * 5;
  float s0 = bias[co], s1 = s0;
  int t0 = 2 * tp, t1 = 2 * tp + 1;
  for (int ci = 0; ci < cfg::CIN; ++ci) {
    const float* xc = xb + ci * cfg::T;
    const float* wk = wc + ci * 5;
#pragma unroll
    for (int k = 0; k < 5; ++k) {
      float wv = wk[k];
      int p0 = t0 + k - 2, p1 = t1 + k - 2;
      if ((unsigned)p0 < (unsigned)cfg::T) s0 += wv * xc[p0];
      if ((unsigned)p1 < (unsigned)cfg::T) s1 += wv * xc[p1];
    }
  }
  h1[idx] = fmaxf(0.f, fmaxf(s0, s1));   // relu/maxpool commute
}

// ---------------------------------------------------------------------------
// Stage 2: conv1d(16->32,k5,same) + relu + maxpool2, writing node features
// directly in node-major bf16: xbf[(b*NEWT+tp)*C2 + co]  (WMMA A operand).
// ---------------------------------------------------------------------------
__global__ void conv2_relu_pool_nodes(const float* __restrict__ h1,
                                      const float* __restrict__ w,
                                      const float* __restrict__ bias,
                                      __hip_bfloat16* __restrict__ xbf) {
  int idx = blockIdx.x * blockDim.x + threadIdx.x;     // B*NEWT*C2 = 2,097,152
  int co = idx & (cfg::C2 - 1);
  int tp = (idx >> 5) & (cfg::NEWT - 1);
  int bb = idx >> 14;
  const float* hb = h1 + (size_t)bb * cfg::C1 * cfg::T1;
  const float* wc = w + (size_t)co * cfg::C1 * 5;
  float s0 = bias[co], s1 = s0;
  int t0 = 2 * tp, t1 = 2 * tp + 1;
  for (int ci = 0; ci < cfg::C1; ++ci) {
    const float* hc = hb + ci * cfg::T1;
    const float* wk = wc + ci * 5;
#pragma unroll
    for (int k = 0; k < 5; ++k) {
      float wv = wk[k];
      int p0 = t0 + k - 2, p1 = t1 + k - 2;
      if ((unsigned)p0 < (unsigned)cfg::T1) s0 += wv * hc[p0];
      if ((unsigned)p1 < (unsigned)cfg::T1) s1 += wv * hc[p1];
    }
  }
  xbf[idx] = __float2bfloat16(fmaxf(0.f, fmaxf(s0, s1)));
}

// ---------------------------------------------------------------------------
// Degree: deg = 1 (self loop) + scatter count at dst; then in-place rsqrt.
// ---------------------------------------------------------------------------
__global__ void deg_fill(float* __restrict__ deg) {
  int i = blockIdx.x * blockDim.x + threadIdx.x;
  if (i < cfg::N) deg[i] = 1.0f;
}
__global__ void deg_count(const int* __restrict__ ei, float* __restrict__ deg) {
  int e = blockIdx.x * blockDim.x + threadIdx.x;
  if (e < cfg::E) atomicAdd(&deg[ei[cfg::E + e]], 1.0f);
}
__global__ void deg_rsqrt(float* __restrict__ deg) {
  int i = blockIdx.x * blockDim.x + threadIdx.x;
  if (i < cfg::N) deg[i] = rsqrtf(deg[i]);
}

// ---------------------------------------------------------------------------
// Pack W[K,HID] (row major f32) into the CDNA5 WMMA bf16 B-operand layout:
// Wp[((c*8 + ntile)*32 + lane)*16 + e] = W[(c*32 + (lane<16?0:8) + (e&7) + (e<8?0:16))*HID
//                                          + ntile*16 + (lane&15)]
// so each lane of the GEMM wave does one contiguous 32B load per K-slice.
// ---------------------------------------------------------------------------
__global__ void pack_w_wmma(const float* __restrict__ W,
                            __hip_bfloat16* __restrict__ Wp) {
  int idx = blockIdx.x * blockDim.x + threadIdx.x;   // (K/32)*8*32*16
  int e    = idx & 15;
  int lane = (idx >> 4) & 31;
  int t    = (idx >> 9) & 7;
  int c    = idx >> 12;
  int n = t * 16 + (lane & 15);
  int kbase = (lane & 16) ? 8 : 0;
  int k = c * 32 + kbase + (e & 7) + ((e & 8) ? 16 : 0);
  Wp[idx] = __float2bfloat16(W[k * cfg::HID + n]);
}

// ---------------------------------------------------------------------------
// GCN dense step: XW = Xbf[N,K] @ Wp  ->  f32 [N,HID] via v_wmma_f32_16x16x32_bf16.
// One wave per 16x16 output tile; K/32 chained WMMAs; all branches wave-uniform
// so EXEC is all-ones at every WMMA.
// ---------------------------------------------------------------------------
__global__ void gcn_gemm_wmma(const __hip_bfloat16* __restrict__ Xbf,
                              const __hip_bfloat16* __restrict__ Wp,
                              float* __restrict__ XW, int Kdim) {
  int gid  = blockIdx.x * blockDim.x + threadIdx.x;
  int lane = gid & 31;
  int wave = gid >> 5;
  int ntile = wave & 7;          // HID/16 = 8 tiles
  int mtile = wave >> 3;         // N/16 tiles
  const __bf16* X = (const __bf16*)Xbf;
  const __bf16* W = (const __bf16*)Wp;

  int row  = mtile * 16 + (lane & 15);       // A: M = lane%16
  int koff = (lane & 16) ? 8 : 0;            // A: upper half-wave K+8
  v8f acc = {};
  int kchunks = Kdim >> 5;
  for (int c = 0; c < kchunks; ++c) {
    const __bf16* pa = X + (size_t)row * Kdim + c * 32 + koff;
    v8bf alo = *(const v8bf*)pa;             // elems 0..7  : K = koff..koff+7
    v8bf ahi = *(const v8bf*)(pa + 16);      // elems 8..15 : K = 16+koff..
    v16bf a = __builtin_shufflevector(alo, ahi,
        0, 1, 2, 3, 4, 5, 6, 7, 8, 9, 10, 11, 12, 13, 14, 15);
    v16bf b = *(const v16bf*)(W + ((size_t)(c * 8 + ntile) * 32 + lane) * 16);
    acc = __builtin_amdgcn_wmma_f32_16x16x32_bf16(
        false, a, false, b, (short)0, acc, false, false);
  }
  // D layout: VGPR r -> M = r + (lane>=16 ? 8 : 0), N = lane&15
  int orow = mtile * 16 + ((lane & 16) ? 8 : 0);
  int col  = ntile * 16 + (lane & 15);
  float* o = XW + (size_t)orow * cfg::HID + col;
#pragma unroll
  for (int r = 0; r < 8; ++r) o[r * cfg::HID] = acc[r];
}

// ---------------------------------------------------------------------------
// agg[n,f] = xw[n,f] * dis[n]^2   (self-loop term, also initializes agg)
// ---------------------------------------------------------------------------
__global__ void init_agg_selfloop(const float* __restrict__ xw,
                                  const float* __restrict__ dis,
                                  float* __restrict__ agg) {
  int idx = blockIdx.x * blockDim.x + threadIdx.x;   // N*HID
  int n = idx >> 7;
  float d = dis[n];
  agg[idx] = xw[idx] * d * d;
}

// ---------------------------------------------------------------------------
// Edge scatter: one wave per edge, lane handles a float4 feature chunk.
// gather xw[src] (global_load_b128) -> 4x global_atomic_add_f32 at dst.
// ---------------------------------------------------------------------------
__global__ void edge_aggregate(const int* __restrict__ ei,
                               const float* __restrict__ dis,
                               const float* __restrict__ xw,
                               float* __restrict__ agg) {
  int gid  = blockIdx.x * blockDim.x + threadIdx.x;  // E*32 exact
  int e    = gid >> 5;
  int lane = gid & 31;
  int s = ei[e];
  int d = ei[cfg::E + e];
  float norm = dis[s] * dis[d];
  const float4 v = *(const float4*)(xw + (size_t)s * cfg::HID + lane * 4);
  float* o = agg + (size_t)d * cfg::HID + lane * 4;
  atomicAdd(o + 0, v.x * norm);
  atomicAdd(o + 1, v.y * norm);
  atomicAdd(o + 2, v.z * norm);
  atomicAdd(o + 3, v.w * norm);
}

// ---------------------------------------------------------------------------
// g = relu(agg + bias); optionally emit bf16 (next layer A) and/or f32 copy.
// ---------------------------------------------------------------------------
__global__ void bias_relu_cast(const float* __restrict__ agg,
                               const float* __restrict__ bias,
                               __hip_bfloat16* __restrict__ obf,
                               float* __restrict__ of32,
                               int write_bf, int write_f32) {
  int idx = blockIdx.x * blockDim.x + threadIdx.x;   // N*HID
  float v = fmaxf(0.f, agg[idx] + bias[idx & (cfg::HID - 1)]);
  if (write_bf)  obf[idx]  = __float2bfloat16(v);
  if (write_f32) of32[idx] = v;
}

// ---------------------------------------------------------------------------
// Mean over 512 nodes per batch row, then 128x6 classifier.
// ---------------------------------------------------------------------------
__global__ void pool_classify(const float* __restrict__ G2,
                              const float* __restrict__ cls_w,
                              const float* __restrict__ cls_b,
                              float* __restrict__ out) {
  __shared__ float sh[cfg::HID];
  int b = blockIdx.x;
  int h = threadIdx.x;
  const float* base = G2 + ((size_t)b * cfg::NEWT) * cfg::HID + h;
  float s = 0.f;
  for (int t = 0; t < cfg::NEWT; ++t) s += base[(size_t)t * cfg::HID];
  sh[h] = s * (1.0f / cfg::NEWT);
  __syncthreads();
  if (h < cfg::OUT) {
    float acc = cls_b[h];
    for (int k = 0; k < cfg::HID; ++k) acc += sh[k] * cls_w[k * cfg::OUT + h];
    out[b * cfg::OUT + h] = acc;
  }
}

// ---------------------------------------------------------------------------
extern "C" void kernel_launch(void* const* d_in, const int* in_sizes, int n_in,
                              void* d_out, int out_size, void* d_ws, size_t ws_size,
                              hipStream_t stream) {
  (void)in_sizes; (void)n_in; (void)out_size; (void)ws_size;

  const float* x       = (const float*)d_in[0];
  const int*   ei      = (const int*)  d_in[1];
  const float* conv1_w = (const float*)d_in[2];
  const float* conv1_b = (const float*)d_in[3];
  const float* conv2_w = (const float*)d_in[4];
  const float* conv2_b = (const float*)d_in[5];
  const float* gcn1_w  = (const float*)d_in[6];
  const float* gcn1_b  = (const float*)d_in[7];
  const float* gcn2_w  = (const float*)d_in[8];
  const float* gcn2_b  = (const float*)d_in[9];
  const float* cls_w   = (const float*)d_in[10];
  const float* cls_b   = (const float*)d_in[11];
  float* out = (float*)d_out;

  // workspace layout (all offsets 256B aligned)
  char* ws = (char*)d_ws;
  size_t off = 0;
  float* h1 = (float*)(ws + off);            off += (size_t)cfg::B * cfg::C1 * cfg::T1 * 4;   //  8 MB
  __hip_bfloat16* Xbf = (__hip_bfloat16*)(ws + off); off += (size_t)cfg::N * cfg::C2 * 2;     //  4 MB
  float* dis = (float*)(ws + off);           off += (size_t)cfg::N * 4;                       // .25 MB
  float* xw  = (float*)(ws + off);           off += (size_t)cfg::N * cfg::HID * 4;            // 32 MB
  float* agg = (float*)(ws + off);           off += (size_t)cfg::N * cfg::HID * 4;            // 32 MB
  __hip_bfloat16* Gbf = (__hip_bfloat16*)(ws + off); off += (size_t)cfg::N * cfg::HID * 2;    // 16 MB
  float* G2  = (float*)(ws + off);           off += (size_t)cfg::N * cfg::HID * 4;            // 32 MB
  __hip_bfloat16* Wp1 = (__hip_bfloat16*)(ws + off); off += (size_t)(cfg::C2  / 32) * 4096 * 2;
  __hip_bfloat16* Wp2 = (__hip_bfloat16*)(ws + off); off += (size_t)(cfg::HID / 32) * 4096 * 2;

  const int TPB = 256;

  // temporal conv stack
  conv1_relu_pool<<<(cfg::B * cfg::C1 * cfg::T1) / TPB, TPB, 0, stream>>>(
      x, conv1_w, conv1_b, h1);
  conv2_relu_pool_nodes<<<(cfg::B * cfg::NEWT * cfg::C2) / TPB, TPB, 0, stream>>>(
      h1, conv2_w, conv2_b, Xbf);

  // degree / normalization
  deg_fill <<<cfg::N / TPB, TPB, 0, stream>>>(dis);
  deg_count<<<cfg::E / TPB, TPB, 0, stream>>>(ei, dis);
  deg_rsqrt<<<cfg::N / TPB, TPB, 0, stream>>>(dis);

  // weight packing for WMMA B operand
  pack_w_wmma<<<((cfg::C2  / 32) * 4096) / TPB, TPB, 0, stream>>>(gcn1_w, Wp1);
  pack_w_wmma<<<((cfg::HID / 32) * 4096) / TPB, TPB, 0, stream>>>(gcn2_w, Wp2);

  const int gemm_threads = (cfg::N / 16) * 8 * 32;   // one wave per 16x16 tile
  const int nh_blocks    = (cfg::N * cfg::HID) / TPB;
  const int edge_blocks  = (cfg::E * 32) / TPB;

  // ---- GCN layer 1 ----
  gcn_gemm_wmma<<<gemm_threads / TPB, TPB, 0, stream>>>(Xbf, Wp1, xw, cfg::C2);
  init_agg_selfloop<<<nh_blocks, TPB, 0, stream>>>(xw, dis, agg);
  edge_aggregate  <<<edge_blocks, TPB, 0, stream>>>(ei, dis, xw, agg);
  bias_relu_cast  <<<nh_blocks, TPB, 0, stream>>>(agg, gcn1_b, Gbf, (float*)0, 1, 0);

  // ---- GCN layer 2 ----
  gcn_gemm_wmma<<<gemm_threads / TPB, TPB, 0, stream>>>(Gbf, Wp2, xw, cfg::HID);
  init_agg_selfloop<<<nh_blocks, TPB, 0, stream>>>(xw, dis, agg);
  edge_aggregate  <<<edge_blocks, TPB, 0, stream>>>(ei, dis, xw, agg);
  bias_relu_cast  <<<nh_blocks, TPB, 0, stream>>>(agg, gcn2_b, (__hip_bfloat16*)0, G2, 0, 1);

  // ---- pool + classifier ----
  pool_classify<<<cfg::B, cfg::HID, 0, stream>>>(G2, cls_w, cls_b, out);
}
